// MHA_28114855920087
// MI455X (gfx1250) — matrix operand order repaired and checked by reading, not compiled, not run on hardware
//
#include <hip/hip_runtime.h>
#include <hip/hip_bf16.h>
#include <math.h>

typedef _Float16 v16h __attribute__((ext_vector_type(16)));
typedef _Float16 v8h  __attribute__((ext_vector_type(8)));
typedef float    v8f  __attribute__((ext_vector_type(8)));

#define HID   1024
#define SEQ   2048
#define BATCH 4
#define HEADS 16
#define HDIM  64
#define ROWS  (BATCH * SEQ)   // 8192

#define MODE_PERHEAD    0   // f16 out, [b,h,s,d]
#define MODE_PERHEAD_T  1   // f16 out, [b,h,d,s]
#define MODE_F32        2   // f32 out, [row, col] (natural)

__device__ __forceinline__ v8f wmma_f16(v16h a, v16h b, v8f c) {
  // D = A(16x32 f16) * B(32x16 f16) + C(16x16 f32)
  return __builtin_amdgcn_wmma_f32_16x16x32_f16(false, a, false, b, (short)0, c,
                                                false, false);
}

// ---- 16-lane reductions: fused DPP row-rotation ops (1 VALU per step) ------
// d = op(row_ror_k(s), s); rotation butterfly k=1,2,4,8 reduces each 16-lane
// DPP row into every lane. VOP2 + DPP16, no LDS traffic, dual-issue friendly.
#define MAXROR(d, s, N)                                                        \
  asm("v_max_num_f32_dpp %0, %1, %1 row_ror:" #N                               \
      " row_mask:0xf bank_mask:0xf bound_ctrl:1"                               \
      : "=v"(d) : "v"(s))
#define ADDROR(d, s, N)                                                        \
  asm("v_add_f32_dpp %0, %1, %1 row_ror:" #N                                   \
      " row_mask:0xf bank_mask:0xf bound_ctrl:1"                               \
      : "=v"(d) : "v"(s))

__device__ __forceinline__ float redmax16(float v) {
  float t;
  MAXROR(t, v, 1);
  MAXROR(v, t, 2);
  MAXROR(t, v, 4);
  MAXROR(v, t, 8);
  return v;
}
__device__ __forceinline__ float redsum16(float v) {
  float t;
  ADDROR(t, v, 1);
  ADDROR(v, t, 2);
  ADDROR(t, v, 4);
  ADDROR(v, t, 8);
  return v;
}

__device__ __forceinline__ v8h cvt8(const float4 a, const float4 b) {
  v8h r = { (_Float16)a.x, (_Float16)a.y, (_Float16)a.z, (_Float16)a.w,
            (_Float16)b.x, (_Float16)b.y, (_Float16)b.z, (_Float16)b.w };
  return r;
}

// ---------------------------------------------------------------------------
// C = A[8192x1024] @ W^T[1024x1024] + bias, tiled 128x128 per workgroup.
// Software-pipelined: global loads for tile k+1 issued before compute of tile
// k; LDS double-buffered (one barrier per k-step). A_F32/MODE are compile-time.
// ---------------------------------------------------------------------------
template <int A_F32, int MODE>
__global__ __launch_bounds__(256)
void mha_gemm_wmma(const void* __restrict__ Araw,
                   const float* __restrict__ W, const float* __restrict__ bias,
                   void* __restrict__ out, float scale)
{
  __shared__ __align__(32) _Float16 a_lds[2][128][32];
  __shared__ __align__(32) _Float16 b_lds[2][128][32];

  const int tid  = threadIdx.x;
  const int lane = tid & 31;
  const int wave = tid >> 5;
  const int g    = lane >> 4;     // half-wave group (0/1)
  const int ln   = lane & 15;
  const int wm   = wave & 3;      // wave tile row (4 x 32 rows)
  const int wn   = wave >> 2;     // wave tile col (2 x 64 cols)
  const int m0   = blockIdx.x * 128;
  const int n0   = blockIdx.y * 128;

  const int kbA = g ? 16 : 0;     // A-matrix per-lane K base
  const int kbB = g ? 16 : 0;     // B-matrix per-lane K base

  const float*    Af = (const float*)Araw;
  const _Float16* Ah = (const _Float16*)Araw;
  const int srow = tid >> 1;          // staging row 0..127
  const int koff = (tid & 1) * 16;    // staging K half

  // staging registers (global -> regs -> LDS pipeline)
  v8h ra0, ra1, rb0, rb1;
  auto gload = [&](int k0) {
    if (A_F32) {
      const float4* p = (const float4*)(Af + (size_t)(m0 + srow) * HID + k0 + koff);
      float4 x0 = p[0], x1 = p[1], x2 = p[2], x3 = p[3];
      ra0 = cvt8(x0, x1); ra1 = cvt8(x2, x3);
    } else {
      const v8h* p = (const v8h*)(Ah + (size_t)(m0 + srow) * HID + k0 + koff);
      ra0 = p[0]; ra1 = p[1];
    }
    const float4* q = (const float4*)(W + (size_t)(n0 + srow) * HID + k0 + koff);
    float4 y0 = q[0], y1 = q[1], y2 = q[2], y3 = q[3];
    rb0 = cvt8(y0, y1); rb1 = cvt8(y2, y3);
  };
  auto lstore = [&](int buf) {
    *(v8h*)&a_lds[buf][srow][koff]     = ra0;
    *(v8h*)&a_lds[buf][srow][koff + 8] = ra1;
    *(v8h*)&b_lds[buf][srow][koff]     = rb0;
    *(v8h*)&b_lds[buf][srow][koff + 8] = rb1;
  };

  const v8f vzero = {0.f,0.f,0.f,0.f,0.f,0.f,0.f,0.f};
  v8f acc[2][4];
#pragma unroll
  for (int mi = 0; mi < 2; mi++)
#pragma unroll
    for (int ni = 0; ni < 4; ni++) acc[mi][ni] = vzero;

  gload(0);
  lstore(0);
  int cur = 0;

  for (int k0 = 0; k0 < HID; k0 += 32) {
    __syncthreads();                       // buf[cur] staged & visible
    const bool more = (k0 + 32 < HID);
    if (more) gload(k0 + 32);              // overlap global latency with WMMA
    if (k0 + 64 < HID) {                   // keep L2 warm two tiles ahead
      if (A_F32)
        __builtin_prefetch((const void*)(Af + (size_t)(m0 + srow) * HID + k0 + 64 + koff), 0, 0);
      else
        __builtin_prefetch((const void*)(Ah + (size_t)(m0 + srow) * HID + k0 + 64 + koff), 0, 0);
      __builtin_prefetch((const void*)(W + (size_t)(n0 + srow) * HID + k0 + 64 + koff), 0, 0);
    }

    // --- A fragments: lane ln holds row, contiguous K [kbA,+16) ---
    v16h afrag[2];
#pragma unroll
    for (int mi = 0; mi < 2; mi++) {
      const _Float16* ap = &a_lds[cur][wm * 32 + mi * 16 + ln][kbA];
      ((v8h*)&afrag[mi])[0] = *(const v8h*)(ap);
      ((v8h*)&afrag[mi])[1] = *(const v8h*)(ap + 8);
    }
    // --- B fragments + WMMA ---
#pragma unroll
    for (int ni = 0; ni < 4; ni++) {
      v16h bfrag = *(const v16h*)&b_lds[cur][wn * 64 + ni * 16 + ln][kbB];
#pragma unroll
      for (int mi = 0; mi < 2; mi++)
        acc[mi][ni] = wmma_f16(afrag[mi], bfrag, acc[mi][ni]);
    }

    if (more) lstore(cur ^ 1);             // after frag loads: no DScnt stall
    cur ^= 1;
  }

  // --- epilogue: bias, scale, layout-specific store (compile-time MODE) ---
#pragma unroll
  for (int mi = 0; mi < 2; mi++) {
#pragma unroll
    for (int ni = 0; ni < 4; ni++) {
      const int col = n0 + wn * 64 + ni * 16 + ln;
      const float bv = bias[col];
#pragma unroll
      for (int r = 0; r < 8; r++) {
        const int row = m0 + wm * 32 + mi * 16 + r + 8 * g;
        float val = acc[mi][ni][r] + bv;
        if (MODE == MODE_F32) {
          ((float*)out)[(size_t)row * HID + col] = val;
        } else {
          val *= scale;
          const int b = row >> 11, s = row & (SEQ - 1);
          const int h = col >> 6,  d = col & (HDIM - 1);
          _Float16* oh = (_Float16*)out;
          if (MODE == MODE_PERHEAD)
            oh[(((size_t)(b * HEADS + h)) * SEQ + s) * HDIM + d] = (_Float16)val;
          else
            oh[(((size_t)(b * HEADS + h)) * HDIM + d) * SEQ + s] = (_Float16)val;
        }
      }
    }
  }
}

// ---------------------------------------------------------------------------
// Flash attention: one WG per (q-block of 128, b*h). 8 waves, each owns 16
// query rows held in registers. KV streamed in 32-wide tiles, LDS double-
// buffered, next tile's global loads overlapped with softmax/WMMA.
// Q pre-scaled by log2(e)/sqrt(HDIM) so softmax uses bare v_exp (2^x).
// V pre-transposed [bh][d][s].
// ---------------------------------------------------------------------------
__global__ __launch_bounds__(256)
void mha_attn_wmma(const _Float16* __restrict__ Q,   // [bh][s][d]
                   const _Float16* __restrict__ K,   // [bh][s][d]
                   const _Float16* __restrict__ VT,  // [bh][d][s]
                   _Float16* __restrict__ O)         // [b*S+s][HID] natural
{
  __shared__ __align__(32) _Float16 k_lds[2][32][64];    // [s_local][d]
  __shared__ __align__(32) _Float16 v_lds[2][64][32];    // [d][s_local]
  __shared__ __align__(32) _Float16 p_lds[8][16][32];    // per-wave P tile

  const int tid  = threadIdx.x;
  const int lane = tid & 31;
  const int wave = tid >> 5;
  const int g    = lane >> 4;
  const int ln   = lane & 15;
  const int qblk = blockIdx.x;   // 0..15
  const int bh   = blockIdx.y;   // 0..63
  const int kbA  = g ? 16 : 0;
  const int kbB  = g ? 16 : 0;

  const _Float16* Qh = Q  + (size_t)bh * SEQ * HDIM;
  const _Float16* Kh = K  + (size_t)bh * SEQ * HDIM;
  const _Float16* Vh = VT + (size_t)bh * HDIM * SEQ;

  // Q fragments for this wave's 16 rows (registers all kernel)
  const int qrow = qblk * 128 + wave * 16 + ln;
  v16h qfrag[2];
#pragma unroll
  for (int ks = 0; ks < 2; ks++) {
    const _Float16* qp = Qh + (size_t)qrow * HDIM + ks * 32 + kbA;
    ((v8h*)&qfrag[ks])[0] = *(const v8h*)(qp);
    ((v8h*)&qfrag[ks])[1] = *(const v8h*)(qp + 8);
  }

  const v8f vzero = {0.f,0.f,0.f,0.f,0.f,0.f,0.f,0.f};
  v8f o_acc[4];
#pragma unroll
  for (int dg = 0; dg < 4; dg++) o_acc[dg] = vzero;
  float mrun[8], lrun[8];
#pragma unroll
  for (int r = 0; r < 8; r++) { mrun[r] = -__builtin_inff(); lrun[r] = 0.f; }

  // cooperative staging (256 threads, one 16B chunk per array each)
  const int kr = tid >> 3, kc = (tid & 7) * 8;   // k tile: 32x64
  const int vr = tid >> 2, vc = (tid & 3) * 8;   // v tile: 64x32

  v8h kreg, vreg;
  auto gload = [&](int s0) {
    kreg = *(const v8h*)(Kh + (size_t)(s0 + kr) * HDIM + kc);
    vreg = *(const v8h*)(Vh + (size_t)vr * SEQ + s0 + vc);
  };
  gload(0);
  int cur = 0;

  for (int s0 = 0; s0 < SEQ; s0 += 32) {
    *(v8h*)&k_lds[cur][kr][kc] = kreg;
    *(v8h*)&v_lds[cur][vr][vc] = vreg;
    __syncthreads();                         // buf[cur] staged & visible
    if (s0 + 32 < SEQ) gload(s0 + 32);       // overlap with scores/softmax
    if (s0 + 64 < SEQ) {
      __builtin_prefetch((const void*)(Kh + (size_t)(s0 + 64 + kr) * HDIM + kc), 0, 0);
      __builtin_prefetch((const void*)(Vh + (size_t)vr * SEQ + s0 + 64 + vc), 0, 0);
    }

    // scores S = Q @ K^T : 16x32 tile (two 16-col WMMA groups, K-dim 64)
    // (already in log2 domain: Q was scaled by log2(e)/sqrt(d))
    v8f sacc[2] = {vzero, vzero};
#pragma unroll
    for (int nt = 0; nt < 2; nt++) {
#pragma unroll
      for (int ks = 0; ks < 2; ks++) {
        v16h bf = *(const v16h*)&k_lds[cur][nt * 16 + ln][ks * 32 + kbB];
        sacc[nt] = wmma_f16(qfrag[ks], bf, sacc[nt]);
      }
    }

    // online softmax in base-2 (fused DPP reductions, bare v_exp via exp2f)
#pragma unroll
    for (int r = 0; r < 8; r++) {
      const float mx   = redmax16(fmaxf(sacc[0][r], sacc[1][r]));
      const float mnew = fmaxf(mrun[r], mx);
      const float alpha = exp2f(mrun[r] - mnew);
      const float p0 = exp2f(sacc[0][r] - mnew);
      const float p1 = exp2f(sacc[1][r] - mnew);
      lrun[r] = lrun[r] * alpha + redsum16(p0 + p1);
      mrun[r] = mnew;
#pragma unroll
      for (int dg = 0; dg < 4; dg++) o_acc[dg][r] *= alpha;
      const int prow = r + 8 * g;
      p_lds[wave][prow][ln]      = (_Float16)p0;
      p_lds[wave][prow][16 + ln] = (_Float16)p1;
    }
    __syncthreads();                         // P relayout visible

    // O += P(16x32) @ V(32x64)
    v16h pfrag;
    {
      const _Float16* pp = &p_lds[wave][ln][kbA];
      ((v8h*)&pfrag)[0] = *(const v8h*)(pp);
      ((v8h*)&pfrag)[1] = *(const v8h*)(pp + 8);
    }
#pragma unroll
    for (int dg = 0; dg < 4; dg++) {
      v16h vf = *(const v16h*)&v_lds[cur][dg * 16 + ln][kbB];
      o_acc[dg] = wmma_f16(pfrag, vf, o_acc[dg]);
    }
    cur ^= 1;
  }

  // normalize + store back to natural [b*S+s][h*64+d] layout (f16 ws)
  const int b = bh >> 4, h = bh & 15;
#pragma unroll
  for (int dg = 0; dg < 4; dg++) {
#pragma unroll
    for (int r = 0; r < 8; r++) {
      const int s = qblk * 128 + wave * 16 + r + 8 * g;
      const float val = o_acc[dg][r] / lrun[r];
      O[((size_t)(b * SEQ + s)) * HID + h * HDIM + dg * 16 + ln] = (_Float16)val;
    }
  }
}

// ---------------------------------------------------------------------------
extern "C" void kernel_launch(void* const* d_in, const int* in_sizes, int n_in,
                              void* d_out, int out_size, void* d_ws, size_t ws_size,
                              hipStream_t stream) {
  (void)in_sizes; (void)n_in; (void)out_size; (void)ws_size;
  const float* X  = (const float*)d_in[0];
  const float* Wq = (const float*)d_in[1];
  const float* bq = (const float*)d_in[2];
  const float* Wk = (const float*)d_in[3];
  const float* bk = (const float*)d_in[4];
  const float* Wv = (const float*)d_in[5];
  const float* bv = (const float*)d_in[6];
  const float* Wo = (const float*)d_in[7];
  const float* bo = (const float*)d_in[8];

  char* ws = (char*)d_ws;
  const size_t SEG = (size_t)ROWS * HID * sizeof(_Float16);  // 16 MB
  _Float16* qh  = (_Float16*)(ws);
  _Float16* kh  = (_Float16*)(ws + SEG);
  _Float16* vth = (_Float16*)(ws + 2 * SEG);
  _Float16* oh  = (_Float16*)(ws + 3 * SEG);

  const dim3 ggrid(ROWS / 128, HID / 128);   // 64 x 8
  const dim3 blk(256);
  // 1/sqrt(64) * log2(e): scores land directly in the base-2 exp domain
  const float qscale = 0.125f * 1.44269504088896340736f;

  mha_gemm_wmma<1, MODE_PERHEAD><<<ggrid, blk, 0, stream>>>(X, Wq, bq, qh,  qscale);
  mha_gemm_wmma<1, MODE_PERHEAD><<<ggrid, blk, 0, stream>>>(X, Wk, bk, kh,  1.0f);
  mha_gemm_wmma<1, MODE_PERHEAD_T><<<ggrid, blk, 0, stream>>>(X, Wv, bv, vth, 1.0f);

  mha_attn_wmma<<<dim3(SEQ / 128, BATCH * HEADS), blk, 0, stream>>>(qh, kh, vth, oh);

  mha_gemm_wmma<0, MODE_F32><<<ggrid, blk, 0, stream>>>(oh, Wo, bo, d_out, 1.0f);
}
